// QEGAN_90280212562553
// MI455X (gfx1250) — matrix operand order repaired and checked by reading, not compiled, not run on hardware
//
#include <hip/hip_runtime.h>
#include <math.h>

typedef __attribute__((ext_vector_type(16))) _Float16 v16h;
typedef __attribute__((ext_vector_type(8)))  float    v8f;

#define PI_HALF 1.5707963267948966f

// ======================================================================
// WMMA GEMM:  C[M x ND] = act( A[M x K] @ W[ND x K]^T + bias )
// A row-major [M,K], W row-major [ND,K]  (both operands K-contiguous).
// Each wave computes a 32x16 output tile (two stacked 16x16 WMMA tiles,
// sharing one B fragment per k-step). blockDim.x = 32 * ceil(ND/16);
// grid.x = M/32.  K, ND, ACT are compile-time so the k-loop fully unrolls.
// ACT: 0 = none, 1 = relu, 2 = tanh(v)*pi/2
// ======================================================================
template <int K, int ND, int ACT>
__global__ void gemm_wmma(const float* __restrict__ A, const float* __restrict__ W,
                          const float* __restrict__ bias, float* __restrict__ C)
{
    const int lane  = threadIdx.x & 31;
    const int ntile = threadIdx.x >> 5;

    const int r15 = lane & 15;
    const int hi  = lane >> 4;             // 0 or 1
    const int row_a0 = blockIdx.x * 32 + r15;       // first 16x16 M-tile
    const int row_a1 = row_a0 + 16;                 // second 16x16 M-tile
    const int row_b  = ntile * 16 + r15;            // W row (N index)
    const int kbA    = hi * 8;             // A: lanes>=16 hold K offsets +8 / +24
    const int kbB    = hi * 8;             // B: lanes>=16 hold K offsets +8 / +24
    const bool bvalid = (row_b < ND);

    v8f acc0 = {};
    v8f acc1 = {};
    #pragma unroll
    for (int kk = 0; kk < K; kk += 32) {
        // ---- B fragment: K = kbB..kbB+7 and kbB+16..kbB+23 of column row_b ----
        v16h bv;
        if (bvalid) {
            const float4* pb = (const float4*)(W + (size_t)row_b * K + kk + kbB);
            float4 b0 = pb[0], b1 = pb[1], b2 = pb[4], b3 = pb[5];
            bv[0]=(_Float16)b0.x;  bv[1]=(_Float16)b0.y;  bv[2]=(_Float16)b0.z;  bv[3]=(_Float16)b0.w;
            bv[4]=(_Float16)b1.x;  bv[5]=(_Float16)b1.y;  bv[6]=(_Float16)b1.z;  bv[7]=(_Float16)b1.w;
            bv[8]=(_Float16)b2.x;  bv[9]=(_Float16)b2.y;  bv[10]=(_Float16)b2.z; bv[11]=(_Float16)b2.w;
            bv[12]=(_Float16)b3.x; bv[13]=(_Float16)b3.y; bv[14]=(_Float16)b3.z; bv[15]=(_Float16)b3.w;
        } else {
            #pragma unroll
            for (int i = 0; i < 16; ++i) bv[i] = (_Float16)0.f;
        }

        // ---- A fragment 0: K = kbA..kbA+7 and kbA+16..kbA+23 of row row_a0 ----
        {
            const float4* pa = (const float4*)(A + (size_t)row_a0 * K + kk + kbA);
            float4 a0 = pa[0], a1 = pa[1], a2 = pa[4], a3 = pa[5];
            v16h av;
            av[0]=(_Float16)a0.x;  av[1]=(_Float16)a0.y;  av[2]=(_Float16)a0.z;  av[3]=(_Float16)a0.w;
            av[4]=(_Float16)a1.x;  av[5]=(_Float16)a1.y;  av[6]=(_Float16)a1.z;  av[7]=(_Float16)a1.w;
            av[8]=(_Float16)a2.x;  av[9]=(_Float16)a2.y;  av[10]=(_Float16)a2.z; av[11]=(_Float16)a2.w;
            av[12]=(_Float16)a3.x; av[13]=(_Float16)a3.y; av[14]=(_Float16)a3.z; av[15]=(_Float16)a3.w;
            acc0 = __builtin_amdgcn_wmma_f32_16x16x32_f16(false, av, false, bv,
                                                          (short)0, acc0, false, false);
        }
        // ---- A fragment 1: same K slice of row row_a1 (B reused) ----
        {
            const float4* pa = (const float4*)(A + (size_t)row_a1 * K + kk + kbA);
            float4 a0 = pa[0], a1 = pa[1], a2 = pa[4], a3 = pa[5];
            v16h av;
            av[0]=(_Float16)a0.x;  av[1]=(_Float16)a0.y;  av[2]=(_Float16)a0.z;  av[3]=(_Float16)a0.w;
            av[4]=(_Float16)a1.x;  av[5]=(_Float16)a1.y;  av[6]=(_Float16)a1.z;  av[7]=(_Float16)a1.w;
            av[8]=(_Float16)a2.x;  av[9]=(_Float16)a2.y;  av[10]=(_Float16)a2.z; av[11]=(_Float16)a2.w;
            av[12]=(_Float16)a3.x; av[13]=(_Float16)a3.y; av[14]=(_Float16)a3.z; av[15]=(_Float16)a3.w;
            acc1 = __builtin_amdgcn_wmma_f32_16x16x32_f16(false, av, false, bv,
                                                          (short)0, acc1, false, false);
        }
    }

    // ---- epilogue: C/D layout: lane n = lane&15, M = r + 8*hi ----
    const int n = ntile * 16 + r15;
    if (n < ND) {
        const float bval = bias[n];
        const int mb0 = blockIdx.x * 32 + hi * 8;
        #pragma unroll
        for (int r = 0; r < 8; ++r) {
            float v = acc0[r] + bval;
            if (ACT == 1)      v = fmaxf(v, 0.f);
            else if (ACT == 2) v = tanhf(v) * PI_HALF;
            C[(size_t)(mb0 + r) * ND + n] = v;
        }
        const int mb1 = mb0 + 16;
        #pragma unroll
        for (int r = 0; r < 8; ++r) {
            float v = acc1[r] + bval;
            if (ACT == 1)      v = fmaxf(v, 0.f);
            else if (ACT == 2) v = tanhf(v) * PI_HALF;
            C[(size_t)(mb1 + r) * ND + n] = v;
        }
    }
}

// ======================================================================
// 4-qubit statevector helpers (16 complex amplitudes per thread).
// wire w <-> bit (3-w) of the flattened index (matches reference ZSIGNS).
// ======================================================================
struct QS { float re[16]; float im[16]; };

__device__ __forceinline__ void qs_init(QS& s) {
    #pragma unroll
    for (int i = 0; i < 16; ++i) { s.re[i] = 0.f; s.im[i] = 0.f; }
    s.re[0] = 1.f;
}

// u = {re00,im00, re01,im01, re10,im10, re11,im11}
__device__ __forceinline__ void apply1(QS& s, int wire, const float* u) {
    const int bit = 1 << (3 - wire);
    #pragma unroll
    for (int i = 0; i < 16; ++i) {
        if (i & bit) continue;
        const int j = i | bit;
        float ar = s.re[i], ai = s.im[i], br = s.re[j], bi = s.im[j];
        s.re[i] = u[0]*ar - u[1]*ai + u[2]*br - u[3]*bi;
        s.im[i] = u[0]*ai + u[1]*ar + u[2]*bi + u[3]*br;
        s.re[j] = u[4]*ar - u[5]*ai + u[6]*br - u[7]*bi;
        s.im[j] = u[4]*ai + u[5]*ar + u[6]*bi + u[7]*br;
    }
}

__device__ __forceinline__ void ry_mat(float th, float* u) {
    float c = __cosf(0.5f * th), sn = __sinf(0.5f * th);
    u[0]=c; u[1]=0.f; u[2]=-sn; u[3]=0.f; u[4]=sn; u[5]=0.f; u[6]=c; u[7]=0.f;
}

__device__ __forceinline__ void had_mat(float* u) {
    const float r = 0.70710678118654752f;
    u[0]=r; u[1]=0.f; u[2]=r; u[3]=0.f; u[4]=r; u[5]=0.f; u[6]=-r; u[7]=0.f;
}

// PennyLane Rot = RZ(omega) RY(theta) RZ(phi)
__device__ __forceinline__ void rot_mat(float phi, float theta, float omega, float* u) {
    float c = __cosf(0.5f * theta), sn = __sinf(0.5f * theta);
    float a = 0.5f * (phi + omega), b = 0.5f * (phi - omega);
    float sa, ca, sb, cb;
    __sincosf(a, &sa, &ca);
    __sincosf(b, &sb, &cb);
    u[0] =  c * ca;  u[1] = -c * sa;   // U00 = e^{-i a} c
    u[2] = -sn * cb; u[3] = -sn * sb;  // U01 = -e^{ i b} s
    u[4] =  sn * cb; u[5] = -sn * sb;  // U10 =  e^{-i b} s
    u[6] =  c * ca;  u[7] =  c * sa;   // U11 =  e^{ i a} c
}

__device__ __forceinline__ void cnot(QS& s, int cw, int tw) {
    const int cb = 1 << (3 - cw), tb = 1 << (3 - tw);
    #pragma unroll
    for (int i = 0; i < 16; ++i) {
        if ((i & cb) && !(i & tb)) {
            const int j = i | tb;
            float tr = s.re[i]; s.re[i] = s.re[j]; s.re[j] = tr;
            float ti = s.im[i]; s.im[i] = s.im[j]; s.im[j] = ti;
        }
    }
}

__device__ __forceinline__ void cry(QS& s, float th, int cw, int tw) {
    const int cb = 1 << (3 - cw), tb = 1 << (3 - tw);
    float c = __cosf(0.5f * th), sn = __sinf(0.5f * th);
    #pragma unroll
    for (int i = 0; i < 16; ++i) {
        if ((i & cb) && !(i & tb)) {
            const int j = i | tb;
            float ar = s.re[i], ai = s.im[i], br = s.re[j], bi = s.im[j];
            s.re[i] = c * ar - sn * br;  s.im[i] = c * ai - sn * bi;
            s.re[j] = sn * ar + c * br;  s.im[j] = sn * ai + c * bi;
        }
    }
}

__device__ __forceinline__ void zexp4(const QS& s, float* z) {
    float e0 = 0.f, e1 = 0.f, e2 = 0.f, e3 = 0.f;
    #pragma unroll
    for (int i = 0; i < 16; ++i) {
        float p = s.re[i]*s.re[i] + s.im[i]*s.im[i];
        e0 += (i & 8) ? -p : p;
        e1 += (i & 4) ? -p : p;
        e2 += (i & 2) ? -p : p;
        e3 += (i & 1) ? -p : p;
    }
    z[0] = e0; z[1] = e1; z[2] = e2; z[3] = e3;
}

// ======================================================================
// entangle circuit: xq[n,4] from h[n,:4] (tanh * pi/2 inside); prm = [2,4,3]
// ======================================================================
__global__ void entangle_kernel(const float* __restrict__ h, const float* __restrict__ prm,
                                float* __restrict__ xq, int Nn)
{
    int n = blockIdx.x * blockDim.x + threadIdx.x;
    if (n >= Nn) return;
    QS st; qs_init(st);
    float u[8];
    #pragma unroll
    for (int i = 0; i < 4; ++i) {
        float th = tanhf(h[(size_t)n * 128 + i]) * PI_HALF;
        ry_mat(th, u); apply1(st, i, u);
    }
    #pragma unroll
    for (int l = 0; l < 2; ++l) {
        cnot(st, 0, 1); cnot(st, 1, 2); cnot(st, 2, 3);
        #pragma unroll
        for (int i = 0; i < 4; ++i) {
            const float* p = prm + ((size_t)l * 4 + i) * 3;
            rot_mat(p[0], p[1], p[2], u); apply1(st, i, u);
        }
        cnot(st, 0, 3);
    }
    float z[4]; zexp4(st, z);
    #pragma unroll
    for (int j = 0; j < 4; ++j) xq[(size_t)n * 4 + j] = z[j];
}

// ======================================================================
// attention circuit per edge; stores exp(score) and accumulates sum-exp.
// prm = [4,3]. Edges: e < E -> (src=ei[e], dst=ei[E+e]); else self-loop.
// ======================================================================
__global__ void attn_kernel(const float* __restrict__ q, const float* __restrict__ k,
                            const float* __restrict__ prm, const int* __restrict__ ei,
                            float* __restrict__ wts, float* __restrict__ sumexp,
                            int E, int Nn)
{
    int e = blockIdx.x * blockDim.x + threadIdx.x;
    if (e >= E + Nn) return;
    int s, d;
    if (e < E) { s = ei[e]; d = ei[E + e]; } else { s = d = e - E; }

    QS st; qs_init(st);
    float u[8];
    #pragma unroll
    for (int i = 0; i < 4; ++i) { ry_mat(q[(size_t)s * 4 + i], u); apply1(st, i, u); }
    had_mat(u);
    #pragma unroll
    for (int i = 0; i < 4; ++i) apply1(st, i, u);
    #pragma unroll
    for (int i = 0; i < 4; ++i) cry(st, k[(size_t)d * 4 + i], i, (i + 1) & 3);
    #pragma unroll
    for (int i = 0; i < 4; ++i) {
        const float* p = prm + i * 3;
        rot_mat(p[0], p[1], p[2], u); apply1(st, i, u);
    }
    float z[4]; zexp4(st, z);
    float score = 0.25f * (z[0] + z[1] + z[2] + z[3]);   // in [-1,1]: exp is safe
    float w = __expf(score);
    wts[e] = w;
    atomicAdd(sumexp, w);
}

// ======================================================================
// aggregation: h_out[dst] += (wts[e]/sumexp) * x_comb[src]; 1 wave per edge.
// ======================================================================
__global__ void agg_kernel(const float* __restrict__ wts, const float* __restrict__ sumexp,
                           const float* __restrict__ xcomb, const int* __restrict__ ei,
                           float* __restrict__ hout, int E, int Nn)
{
    int gid  = blockIdx.x * blockDim.x + threadIdx.x;
    int e    = gid >> 5;
    int lane = gid & 31;
    if (e >= E + Nn) return;
    int s, d;
    if (e < E) { s = ei[e]; d = ei[E + e]; } else { s = d = e - E; }
    float coeff = wts[e] / sumexp[0];
    const float* xs = xcomb + (size_t)s * 128;
    float*       hd = hout  + (size_t)d * 128;
    #pragma unroll
    for (int f = lane; f < 128; f += 32)
        atomicAdd(&hd[f], coeff * xs[f]);
}

// ======================================================================
// path circuit: z4[n,4] from pq[n,4] (already tanh*pi/2); prm = [3,4,3]
// ======================================================================
__global__ void pathz_kernel(const float* __restrict__ pq, const float* __restrict__ prm,
                             float* __restrict__ z4, int Nn)
{
    int n = blockIdx.x * blockDim.x + threadIdx.x;
    if (n >= Nn) return;
    QS st; qs_init(st);
    float u[8];
    had_mat(u);
    #pragma unroll
    for (int i = 0; i < 4; ++i) apply1(st, i, u);
    #pragma unroll
    for (int i = 0; i < 4; ++i) { ry_mat(pq[(size_t)n * 4 + i], u); apply1(st, i, u); }
    #pragma unroll
    for (int l = 0; l < 3; ++l) {
        #pragma unroll
        for (int i = 0; i < 4; ++i) {
            const float* p = prm + ((size_t)l * 4 + i) * 3;
            rot_mat(p[0], p[1], p[2], u); apply1(st, i, u);
        }
        cnot(st, 0, 1); cnot(st, 1, 2); cnot(st, 2, 3);
    }
    float z[4]; zexp4(st, z);
    #pragma unroll
    for (int j = 0; j < 4; ++j) z4[(size_t)n * 4 + j] = z[j];
}

// ======================================================================
// out[m,f] (+)= b[f] + sum_j z4[m,j] * Wp[f,j]     (Wp row-major [128,4])
// ======================================================================
__global__ void proj4_kernel(const float* __restrict__ z4, const float* __restrict__ Wp,
                             const float* __restrict__ bp, float* __restrict__ out,
                             int Nn, int accum)
{
    int idx = blockIdx.x * blockDim.x + threadIdx.x;
    if (idx >= Nn * 128) return;
    int m = idx >> 7, f = idx & 127;
    const float* z = z4 + (size_t)m * 4;
    const float* w = Wp + (size_t)f * 4;
    float v = bp[f] + z[0]*w[0] + z[1]*w[1] + z[2]*w[2] + z[3]*w[3];
    if (accum) out[idx] += v; else out[idx] = v;
}

__global__ void zero_kernel(float* __restrict__ p, int n) {
    int i = blockIdx.x * blockDim.x + threadIdx.x;
    if (i < n) p[i] = 0.f;
}

__global__ void relu_kernel(float* __restrict__ p, int n) {
    int i = blockIdx.x * blockDim.x + threadIdx.x;
    if (i < n) p[i] = fmaxf(p[i], 0.f);
}

// ======================================================================
extern "C" void kernel_launch(void* const* d_in, const int* in_sizes, int n_in,
                              void* d_out, int out_size, void* d_ws, size_t ws_size,
                              hipStream_t stream)
{
    const float* x       = (const float*)d_in[0];
    const float* W_in    = (const float*)d_in[1];
    const float* b_in    = (const float*)d_in[2];
    const float* lin_W   = (const float*)d_in[3];
    const float* lin_b   = (const float*)d_in[4];
    const float* qproj_W = (const float*)d_in[5];
    const float* qproj_b = (const float*)d_in[6];
    const float* ent_p   = (const float*)d_in[7];
    const float* attq_W  = (const float*)d_in[8];
    const float* attq_b  = (const float*)d_in[9];
    const float* attk_W  = (const float*)d_in[10];
    const float* attk_b  = (const float*)d_in[11];
    const float* att_qp  = (const float*)d_in[12];
    const float* path_p  = (const float*)d_in[13];
    const float* pin_W   = (const float*)d_in[14];
    const float* pin_b   = (const float*)d_in[15];
    const float* pout_W  = (const float*)d_in[16];
    const float* pout_b  = (const float*)d_in[17];
    const float* out_W   = (const float*)d_in[18];
    const float* out_b   = (const float*)d_in[19];
    const int*   eidx    = (const int*)d_in[20];

    const int Nn = in_sizes[0] / 64;    // 20000
    const int E  = in_sizes[20] / 2;    // 300000
    const int DH = 128, DOUT = 32, L = 2;
    (void)n_in; (void)out_size; (void)ws_size; (void)DOUT;

    // workspace carve-up (256B aligned)
    char* ws = (char*)d_ws;
    size_t off = 0;
    auto carve = [&](size_t bytes) -> float* {
        float* p = (float*)(ws + off);
        off += (bytes + 255) & ~(size_t)255;
        return p;
    };
    float* hA    = carve((size_t)Nn * DH * 4);
    float* hB    = carve((size_t)Nn * DH * 4);
    float* xcomb = carve((size_t)Nn * DH * 4);
    float* xq    = carve((size_t)Nn * 4 * 4);
    float* qb    = carve((size_t)Nn * 4 * 4);
    float* kb    = carve((size_t)Nn * 4 * 4);
    float* wts   = carve((size_t)(E + Nn) * 4);
    float* sume  = carve(256);

    const int MT2  = Nn / 32;                 // 625 blocks (32 rows each)
    const int NH   = Nn * DH;                 // 2.56M
    const int TOTE = E + Nn;                  // 320k edges incl. self loops

    // h = relu(x @ W_in^T + b_in)          [K=64, N=128]
    gemm_wmma<64, 128, 1><<<MT2, 256, 0, stream>>>(x, W_in, b_in, hA);

    float* hcur = hA;
    float* hnxt = hB;
    for (int l = 0; l < L; ++l) {
        // x_cls = h @ lin_W^T + lin_b       [K=128, N=128]
        gemm_wmma<128, 128, 0><<<MT2, 256, 0, stream>>>(hcur, lin_W + (size_t)l * DH * DH,
                                                        lin_b + l * DH, xcomb);
        // xq = entangle(tanh(h[:, :4]) * pi/2)
        entangle_kernel<<<(Nn + 127) / 128, 128, 0, stream>>>(hcur, ent_p + l * 24, xq, Nn);
        // x_comb = x_cls + xq @ qproj_W^T + qproj_b
        proj4_kernel<<<(NH + 255) / 256, 256, 0, stream>>>(xq, qproj_W + (size_t)l * DH * 4,
                                                           qproj_b + l * DH, xcomb, Nn, 1);
        // q, k = tanh(h @ W^T + b) * pi/2   [K=128, N=4 zero-padded tile]
        gemm_wmma<128, 4, 2><<<MT2, 32, 0, stream>>>(hcur, attq_W + (size_t)l * 4 * DH,
                                                     attq_b + l * 4, qb);
        gemm_wmma<128, 4, 2><<<MT2, 32, 0, stream>>>(hcur, attk_W + (size_t)l * 4 * DH,
                                                     attk_b + l * 4, kb);
        // edge scores + global softmax denominator
        zero_kernel<<<1, 32, 0, stream>>>(sume, 1);
        attn_kernel<<<(TOTE + 127) / 128, 128, 0, stream>>>(qb, kb, att_qp + l * 12,
                                                            eidx, wts, sume, E, Nn);
        // h_next = relu( segment_sum( att * x_comb[src] -> dst ) )
        zero_kernel<<<(NH + 255) / 256, 256, 0, stream>>>(hnxt, NH);
        agg_kernel<<<((size_t)TOTE * 32 + 255) / 256, 256, 0, stream>>>(wts, sume, xcomb,
                                                                        eidx, hnxt, E, Nn);
        relu_kernel<<<(NH + 255) / 256, 256, 0, stream>>>(hnxt, NH);
        float* t = hcur; hcur = hnxt; hnxt = t;
    }

    // pq = tanh(h @ path_in_W^T + b) * pi/2   [K=128, N=4]
    gemm_wmma<128, 4, 2><<<MT2, 32, 0, stream>>>(hcur, pin_W, pin_b, qb);
    // z = path_circuit(pq); h = z @ path_out_W^T + path_out_b
    pathz_kernel<<<(Nn + 127) / 128, 128, 0, stream>>>(qb, path_p, xq, Nn);
    proj4_kernel<<<(NH + 255) / 256, 256, 0, stream>>>(xq, pout_W, pout_b, hnxt, Nn, 0);
    // out = h @ out_W^T + out_b  -> d_out [Nn, 32]   [K=128, N=32]
    gemm_wmma<128, 32, 0><<<MT2, 64, 0, stream>>>(hnxt, out_W, out_b, (float*)d_out);
}
